// BitLinear_46780783788344
// MI455X (gfx1250) — compile-verified, matrix operand order checked by reading, not run on hardware
//
#include <hip/hip_runtime.h>
#include <hip/hip_bf16.h>

// ---------------- types for WMMA ----------------
typedef __attribute__((ext_vector_type(16))) __bf16 v16bf;
typedef __attribute__((ext_vector_type(8)))  float  v8f;
typedef __attribute__((ext_vector_type(4)))  int    v4i;

union FragAB { v16bf v; uint4 q[2]; };

// ---------------- problem sizes ----------------
#define M_DIM 8192            // 4 * 2048
#define N_DIM 4096
#define K_DIM 4096
#define NWEIGHT (N_DIM * K_DIM)       // 16,777,216
#define K_RANK  8388608u              // int(n * 0.5), 1-indexed k-th smallest |w|

// ---------------- workspace layout (bytes) ----------------
#define HIST_HI_OFF 0u
#define HIST_LO_OFF (1u << 18)              // 262144
#define PSUM_OFF    (1u << 19)              // 524288  (512 floats)
#define PCNT_OFF    (PSUM_OFF + 2048u)      // 512 u32
#define CTRL_OFF    (PSUM_OFF + 4096u)      // 528384
// ctrl words: [0]=rank0 [1]=binhi [2]=rank1 [3]=binlo [4]=rank2 [5]=thr bits [6]=scale
#define CTRL_WORD0_IDX (CTRL_OFF / 4u)                   // 132096
#define INIT_WORDS     (CTRL_WORD0_IDX + 16u)            // zero everything up to ctrl end
#define WQ_OFF      (1u << 20)                            // bf16 ternary weight, 32 MB
#define XB_OFF      (WQ_OFF + (size_t)NWEIGHT * 2u)       // bf16 x, 64 MB

// ---------------- GEMM tiling ----------------
#define MT 128
#define NT 64
#define KT 64
#define KTILES (K_DIM / KT)   // 64
#define LDA 72    // padded LDS row stride (elements); 144B rows keep 16B alignment

// ---------------- CDNA5 async global->LDS path (probe via __has_builtin) -------------
#if defined(__HIP_DEVICE_COMPILE__) && \
    __has_builtin(__builtin_amdgcn_global_load_async_to_lds_b128) && \
    __has_builtin(__builtin_amdgcn_s_wait_asynccnt)
#define USE_ASYNC_LDS 1
__device__ __forceinline__ void async_cp16(const void* g, void* l) {
    // param0: v4i in global (AS1); param1: v4i in LDS (AS3); C-style casts lower to
    // proper addrspacecast (needed for generic->AS3 aperture translation).
    __builtin_amdgcn_global_load_async_to_lds_b128(
        (__attribute__((address_space(1))) v4i*)g,
        (__attribute__((address_space(3))) v4i*)l, 0, 0);
}
#else
#define USE_ASYNC_LDS 0
#endif

__device__ __forceinline__ unsigned short f2bf(float f) {
    unsigned u = __float_as_uint(f);
    return (unsigned short)((u + 0x7FFFu + ((u >> 16) & 1u)) >> 16);  // RNE
}

// ---------------- 0) zero control region, seed rank0 ----------------
__global__ void k_init(unsigned* ws) {
    unsigned i = blockIdx.x * blockDim.x + threadIdx.x;
    if (i < INIT_WORDS) ws[i] = (i == CTRL_WORD0_IDX) ? K_RANK : 0u;
}

// ---------------- 1) histogram of hi 16 bits of |w| ----------------
__global__ void k_hist_hi(const float* __restrict__ w, unsigned* __restrict__ hist) {
    int i = blockIdx.x * blockDim.x + threadIdx.x;
    int stride = gridDim.x * blockDim.x;
    for (; i < NWEIGHT; i += stride) {
        unsigned b = __float_as_uint(fabsf(w[i])) >> 16;
        atomicAdd(&hist[b], 1u);
    }
}

// ---------------- 2) select bin containing rank (shared by hi/lo passes) ----------------
__global__ void k_select(const unsigned* __restrict__ hist, const unsigned* __restrict__ rank_in,
                         unsigned* __restrict__ bin_out, unsigned* __restrict__ rem_out) {
    __shared__ unsigned part[1024];
    unsigned t = threadIdx.x;
    unsigned s = 0;
#pragma unroll 8
    for (int j = 0; j < 64; ++j) s += hist[t * 64 + j];
    part[t] = s;
    __syncthreads();
    if (t == 0) {
        unsigned k = *rank_in;
        unsigned c = 0, seg = 0;
        for (seg = 0; seg < 1023; ++seg) {
            if (c + part[seg] >= k) break;
            c += part[seg];
        }
        unsigned bin = seg * 64, rem = 1;
        for (unsigned j = 0; j < 64; ++j) {
            unsigned nc = c + hist[seg * 64 + j];
            if (nc >= k) { bin = seg * 64 + j; rem = k - c; break; }
            c = nc;
        }
        *bin_out = bin;
        *rem_out = rem;
    }
}

// ---------------- 3) histogram of lo 16 bits within the selected hi bin ----------------
__global__ void k_hist_lo(const float* __restrict__ w, unsigned* __restrict__ hist,
                          const unsigned* __restrict__ binhi) {
    unsigned B = *binhi;
    int i = blockIdx.x * blockDim.x + threadIdx.x;
    int stride = gridDim.x * blockDim.x;
    for (; i < NWEIGHT; i += stride) {
        unsigned bits = __float_as_uint(fabsf(w[i]));
        if ((bits >> 16) == B) atomicAdd(&hist[bits & 0xFFFFu], 1u);
    }
}

// ---------------- 4) assemble exact threshold bits ----------------
__global__ void k_thresh(unsigned* ctrl) {
    ctrl[5] = (ctrl[1] << 16) | ctrl[3];
}

// ---------------- 5) ternary quantize; deterministic partial sums ----------------
__global__ void k_quantize(const float* __restrict__ w, unsigned short* __restrict__ wq,
                           const unsigned* __restrict__ ctrl,
                           float* __restrict__ psum, unsigned* __restrict__ pcnt) {
    float thr = __uint_as_float(ctrl[5]);
    float s = 0.0f;
    unsigned c = 0;
    int i = blockIdx.x * blockDim.x + threadIdx.x;
    int stride = gridDim.x * blockDim.x;
    for (; i < NWEIGHT; i += stride) {
        float v = w[i];
        float a = fabsf(v);
        unsigned short q = 0;
        if (a > thr) {                      // strict >, matching reference mask
            q = (v > 0.0f) ? (unsigned short)0x3F80 : (unsigned short)0xBF80;
            s += a;
            ++c;
        }
        wq[i] = q;
    }
    __shared__ float ls[256];
    __shared__ unsigned lc[256];
    int t = threadIdx.x;
    ls[t] = s; lc[t] = c;
    __syncthreads();
    for (int d = 128; d > 0; d >>= 1) {
        if (t < d) { ls[t] += ls[t + d]; lc[t] += lc[t + d]; }
        __syncthreads();
    }
    if (t == 0) { psum[blockIdx.x] = ls[0]; pcnt[blockIdx.x] = lc[0]; }
}

// ---------------- 6) scale = sum / max(count,1), fixed-order tree ----------------
__global__ void k_scale(const float* __restrict__ psum, const unsigned* __restrict__ pcnt,
                        float* __restrict__ scale_out) {
    __shared__ float fs[256];
    __shared__ unsigned cs[256];
    int t = threadIdx.x;
    fs[t] = psum[t] + psum[t + 256];
    cs[t] = pcnt[t] + pcnt[t + 256];
    __syncthreads();
    for (int d = 128; d > 0; d >>= 1) {
        if (t < d) { fs[t] += fs[t + d]; cs[t] += cs[t + d]; }
        __syncthreads();
    }
    if (t == 0) {
        unsigned cnt = cs[0] ? cs[0] : 1u;
        *scale_out = fs[0] / (float)cnt;
    }
}

// ---------------- 7) x fp32 -> bf16 (RNE), 4-wide ----------------
__global__ void k_xcvt(const float* __restrict__ x, unsigned short* __restrict__ xb) {
    int i = blockIdx.x * blockDim.x + threadIdx.x;   // one float4 per thread
    float4 f = ((const float4*)x)[i];
    ushort4 o;
    o.x = f2bf(f.x); o.y = f2bf(f.y); o.z = f2bf(f.z); o.w = f2bf(f.w);
    ((ushort4*)xb)[i] = o;
}

// ---------------- 8) bf16 WMMA GEMM: out = scale * (xb @ wq^T) + bias ----------------
__global__ __launch_bounds__(256, 1) void k_gemm(
    const unsigned short* __restrict__ xb,   // [M_DIM][K_DIM] bf16
    const unsigned short* __restrict__ wq,   // [N_DIM][K_DIM] bf16 ternary
    const float* __restrict__ bias,
    const float* __restrict__ scale_p,
    float* __restrict__ out) {

    __shared__ unsigned short xs[2][MT][LDA];
    __shared__ unsigned short wt[2][NT][LDA];

    const int tid = threadIdx.x;
    const int m0 = blockIdx.y * MT;
    const int n0 = blockIdx.x * NT;

    // cooperative tile load mapping: 8 threads per 64-element row chunk (128B)
    const int lrow = tid >> 3;          // 0..31
    const int lcol = (tid & 7) * 8;     // element offset, 16B granules

    // wave/fragment mapping
    const int lane = tid & 31;
    const int wid  = tid >> 5;          // 0..7
    const int wm   = (wid & 3) * 32;    // wave's M sub-tile base in block tile
    const int wn   = (wid >> 2) * 32;   // wave's N sub-tile base
    const int frow = lane & 15;         // row within 16x16
    const int fk   = (lane >> 4) * 8;   // K phase per ISA 16-bit A layout

    v8f acc[2][2];
#pragma unroll
    for (int i = 0; i < 2; ++i)
#pragma unroll
        for (int j = 0; j < 2; ++j)
            acc[i][j] = (v8f){0.f, 0.f, 0.f, 0.f, 0.f, 0.f, 0.f, 0.f};

    // compute one KT tile out of LDS buffer `buf`
    auto compute = [&](int buf) {
#pragma unroll
        for (int ks = 0; ks < KT; ks += 32) {
            FragAB a0, a1, b0, b1;
            a0.q[0] = *(const uint4*)&xs[buf][wm + frow][ks + fk];
            a0.q[1] = *(const uint4*)&xs[buf][wm + frow][ks + fk + 16];
            a1.q[0] = *(const uint4*)&xs[buf][wm + 16 + frow][ks + fk];
            a1.q[1] = *(const uint4*)&xs[buf][wm + 16 + frow][ks + fk + 16];
            b0.q[0] = *(const uint4*)&wt[buf][wn + frow][ks + fk];
            b0.q[1] = *(const uint4*)&wt[buf][wn + frow][ks + fk + 16];
            b1.q[0] = *(const uint4*)&wt[buf][wn + 16 + frow][ks + fk];
            b1.q[1] = *(const uint4*)&wt[buf][wn + 16 + frow][ks + fk + 16];

            acc[0][0] = __builtin_amdgcn_wmma_f32_16x16x32_bf16(false, a0.v, false, b0.v,
                                                                (short)0, acc[0][0], false, false);
            acc[0][1] = __builtin_amdgcn_wmma_f32_16x16x32_bf16(false, a0.v, false, b1.v,
                                                                (short)0, acc[0][1], false, false);
            acc[1][0] = __builtin_amdgcn_wmma_f32_16x16x32_bf16(false, a1.v, false, b0.v,
                                                                (short)0, acc[1][0], false, false);
            acc[1][1] = __builtin_amdgcn_wmma_f32_16x16x32_bf16(false, a1.v, false, b1.v,
                                                                (short)0, acc[1][1], false, false);
        }
    };

#if USE_ASYNC_LDS
    // --- async DMA pipeline: memory -> LDS without touching VGPRs (ASYNCcnt) ---
    auto issue_tile = [&](int buf, int kbase) {
#pragma unroll
        for (int r = 0; r < 4; ++r)
            async_cp16(&xb[(size_t)(m0 + lrow + r * 32) * K_DIM + kbase + lcol],
                       &xs[buf][lrow + r * 32][lcol]);
#pragma unroll
        for (int r = 0; r < 2; ++r)
            async_cp16(&wq[(size_t)(n0 + lrow + r * 32) * K_DIM + kbase + lcol],
                       &wt[buf][lrow + r * 32][lcol]);
    };

    issue_tile(0, 0);                         // 6 async ops in flight
    for (int kt = 0; kt < KTILES - 1; ++kt) {
        const int cur = kt & 1;
        issue_tile(cur ^ 1, (kt + 1) * KT);   // 12 in flight
        __builtin_amdgcn_s_wait_asynccnt(6);  // in-order: current tile landed
        __syncthreads();                      // all waves' data visible
        compute(cur);
        __syncthreads();                      // done reading cur before iter kt+1 refills it
    }
    __builtin_amdgcn_s_wait_asynccnt(0);
    __syncthreads();
    compute((KTILES - 1) & 1);
#else
    // --- fallback: synchronous single-buffer staging through registers ---
    for (int kt = 0; kt < KTILES; ++kt) {
        const int kbase = kt * KT;
        uint4 xr[4], wr[2];
#pragma unroll
        for (int r = 0; r < 4; ++r)
            xr[r] = *(const uint4*)&xb[(size_t)(m0 + lrow + r * 32) * K_DIM + kbase + lcol];
#pragma unroll
        for (int r = 0; r < 2; ++r)
            wr[r] = *(const uint4*)&wq[(size_t)(n0 + lrow + r * 32) * K_DIM + kbase + lcol];
        __syncthreads();                      // previous compute done before overwrite
#pragma unroll
        for (int r = 0; r < 4; ++r)
            *(uint4*)&xs[0][lrow + r * 32][lcol] = xr[r];
#pragma unroll
        for (int r = 0; r < 2; ++r)
            *(uint4*)&wt[0][lrow + r * 32][lcol] = wr[r];
        __syncthreads();
        compute(0);
    }
#endif

    // epilogue: C/D layout — VGPR r holds M=r (lanes 0-15) / M=8+r (lanes 16-31), N=lane&15
    const float scale = *scale_p;
#pragma unroll
    for (int i = 0; i < 2; ++i) {
#pragma unroll
        for (int j = 0; j < 2; ++j) {
            const int ncol = n0 + wn + j * 16 + (lane & 15);
            const float b = bias[ncol];
#pragma unroll
            for (int r = 0; r < 8; ++r) {
                const int mrow = m0 + wm + i * 16 + (lane >> 4) * 8 + r;
                out[(size_t)mrow * N_DIM + ncol] = acc[i][j][r] * scale + b;
            }
        }
    }
}

// ---------------- host-side launcher ----------------
extern "C" void kernel_launch(void* const* d_in, const int* in_sizes, int n_in,
                              void* d_out, int out_size, void* d_ws, size_t ws_size,
                              hipStream_t stream) {
    const float* x    = (const float*)d_in[0];
    const float* w    = (const float*)d_in[1];
    const float* bias = (const float*)d_in[2];
    float* out = (float*)d_out;
    char* ws = (char*)d_ws;

    unsigned*       hist_hi = (unsigned*)(ws + HIST_HI_OFF);
    unsigned*       hist_lo = (unsigned*)(ws + HIST_LO_OFF);
    float*          psum    = (float*)(ws + PSUM_OFF);
    unsigned*       pcnt    = (unsigned*)(ws + PCNT_OFF);
    unsigned*       ctrl    = (unsigned*)(ws + CTRL_OFF);
    unsigned short* wq      = (unsigned short*)(ws + WQ_OFF);
    unsigned short* xb      = (unsigned short*)(ws + XB_OFF);

    // 0) zero histograms/partials/ctrl; seed rank0 = K_RANK
    k_init<<<(INIT_WORDS + 255) / 256, 256, 0, stream>>>((unsigned*)ws);
    // 1) hi-16 histogram of |w| bit patterns
    k_hist_hi<<<1024, 256, 0, stream>>>(w, hist_hi);
    // 2) select hi bin + residual rank
    k_select<<<1, 1024, 0, stream>>>(hist_hi, &ctrl[0], &ctrl[1], &ctrl[2]);
    // 3) lo-16 histogram within selected bin
    k_hist_lo<<<1024, 256, 0, stream>>>(w, hist_lo, &ctrl[1]);
    // 4) select lo bin -> exact threshold
    k_select<<<1, 1024, 0, stream>>>(hist_lo, &ctrl[2], &ctrl[3], &ctrl[4]);
    k_thresh<<<1, 1, 0, stream>>>(ctrl);
    // 5) ternary quantize + deterministic partial reduction (512 blocks fixed)
    k_quantize<<<512, 256, 0, stream>>>(w, wq, ctrl, psum, pcnt);
    // 6) scale
    k_scale<<<1, 256, 0, stream>>>(psum, pcnt, (float*)&ctrl[6]);
    // 7) x -> bf16
    k_xcvt<<<(M_DIM * K_DIM / 4) / 256, 256, 0, stream>>>(x, xb);
    // 8) WMMA GEMM + bias epilogue
    dim3 grid(N_DIM / NT, M_DIM / MT);
    k_gemm<<<grid, 256, 0, stream>>>(xb, wq, bias, (const float*)&ctrl[6], out);
    (void)in_sizes; (void)n_in; (void)out_size; (void)ws_size;
}
